// WMoE_33655363731933
// MI455X (gfx1250) — matrix operand order repaired and checked by reading, not compiled, not run on hardware
//
#include <hip/hip_runtime.h>
#include <hip/hip_bf16.h>

// ---------------------------------------------------------------------------
// CDNA5 (gfx1250) fused conv+MoE pipeline, bf16 WMMA everywhere.
// ---------------------------------------------------------------------------

typedef __attribute__((ext_vector_type(16))) __bf16 v16bf;
typedef __attribute__((ext_vector_type(8)))  float  v8f;

#define HWPIX 65536   // 256*256

__device__ __forceinline__ unsigned short f2bf(float f) {
  unsigned u = __float_as_uint(f);
  unsigned r = (u + 0x7FFFu + ((u >> 16) & 1u)) >> 16;   // RNE
  return (unsigned short)r;
}
__device__ __forceinline__ float bf2f(unsigned short h) {
  return __uint_as_float(((unsigned)h) << 16);
}
__device__ __forceinline__ unsigned packbf(float lo, float hi) {
  return (unsigned)f2bf(lo) | ((unsigned)f2bf(hi) << 16);
}
__device__ __forceinline__ float gelu_exact(float x) {
  return 0.5f * x * (1.0f + erff(x * 0.70710678118654752f));
}
__device__ __forceinline__ v8f wmma_bf16(v16bf a, v16bf b, v8f c) {
  return __builtin_amdgcn_wmma_f32_16x16x32_bf16(false, a, false, b, (short)0, c,
                                                 false, false);
}

// ---- workspace layout (bytes) ---------------------------------------------
// A-fragments: 104 frags * 1KB (32 lanes * 8 dwords)
//   c1  : frag  0..71  (m 0..3  x kc 0..17)
//   c2  : frag 72..87  (m 0..7  x kc 0..1)
//   proj: frag 88..95  (m 0..3  x kc 0..1)
//   W1all: 96..97, W2all: 98..99, W3all: 100..103 (m 0..3, single K-chunk)
static constexpr size_t AFRAG_OFF = 0;                        // 106496 B
static constexpr size_t B1_OFF    = 106496;                   // 16 f32
static constexpr size_t B2_OFF    = 106560;                   // 16 f32
static constexpr size_t MASK_OFF  = 106624;                   // 4*16 f32
static constexpr size_t B3M_OFF   = 106880;                   // 4*64 f32
static constexpr size_t XA_OFF    = 110592;                   // bf16 [4][65536][64]
static constexpr size_t KW_OFF    = XA_OFF + 33554432ull;     // bf16 [4][65536][64]
static constexpr size_t XS_OFF    = KW_OFF + 33554432ull;     // bf16 [4][65536][64]
static constexpr size_t PART_OFF  = XS_OFF + 33554432ull;     // f32 [4][1024][64]

// ===========================================================================
// K0: repack weights into per-lane WMMA A-fragments (bf16) + stacked biases.
// A-frag 16x32 bf16 k order used here (contiguous; B-frags match): lane L -> M=L%16, VGPR v half hi:
//   K = (L/16)*16 + v*2 + hi
// ===========================================================================
__global__ void __launch_bounds__(256)
k0_prep(const float* c1w, const float* c2w, const float* projw,
        const float* e0w1, const float* e0b1, const float* e0w2, const float* e0b2, const float* e0w3,
        const float* e1w1, const float* e1b1, const float* e1w2, const float* e1b2, const float* e1w3,
        const float* e2w1, const float* e2b1, const float* e2w2, const float* e2b2, const float* e2w3,
        unsigned* afr, float* b1all, float* b2all) {
  int tid = threadIdx.x;
  for (int it = tid; it < 104 * 32; it += 256) {
    int f = it >> 5, lane = it & 31;
    int M = lane & 15, lh = lane >> 4;
    unsigned* dst = afr + f * 256 + lane * 8;
    for (int v = 0; v < 8; ++v) {
      float pv[2];
      for (int hi = 0; hi < 2; ++hi) {
        int K = (lh << 4) + (v << 1) + hi;
        float x = 0.0f;
        if (f < 72) {                       // conv1: K-chunk = (tap, 32-ch half)
          int m = f / 18, kc = f % 18;
          int tap = kc >> 1, c = ((kc & 1) << 5) + K, o = m * 16 + M;
          x = c1w[(o * 64 + c) * 9 + tap];
        } else if (f < 88) {                // c2 expand [128][64]
          int g = f - 72, m = g >> 1, kc = g & 1;
          x = c2w[(m * 16 + M) * 64 + (kc << 5) + K];
        } else if (f < 96) {                // proj [64][64]
          int g = f - 88, m = g >> 1, kc = g & 1;
          x = projw[(m * 16 + M) * 64 + (kc << 5) + K];
        } else if (f < 98) {                // W1all: rows 0-1 e0, 2-5 e1, 6-13 e2
          int c = ((f - 96) << 5) + K;
          if (M < 2)       x = e0w1[M * 64 + c];
          else if (M < 6)  x = e1w1[(M - 2) * 64 + c];
          else if (M < 14) x = e2w1[(M - 6) * 64 + c];
        } else if (f < 100) {               // W2all
          int c = ((f - 98) << 5) + K;
          if (M < 2)       x = e0w2[M * 64 + c];
          else if (M < 6)  x = e1w2[(M - 2) * 64 + c];
          else if (M < 14) x = e2w2[(M - 6) * 64 + c];
        } else {                            // W3all: [64][K=row, pad to 32]
          int o = (f - 100) * 16 + M;
          if (K < 2)       x = e0w3[o * 2 + K];
          else if (K < 6)  x = e1w3[o * 4 + (K - 2)];
          else if (K < 14) x = e2w3[o * 8 + (K - 6)];
        }
        pv[hi] = x;
      }
      dst[v] = packbf(pv[0], pv[1]);
    }
  }
  if (tid < 16) {
    float b1 = tid < 2 ? e0b1[tid] : tid < 6 ? e1b1[tid - 2] : tid < 14 ? e2b1[tid - 6] : 0.0f;
    float b2 = tid < 2 ? e0b2[tid] : tid < 6 ? e1b2[tid - 2] : tid < 14 ? e2b2[tid - 6] : 0.0f;
    b1all[tid] = b1;
    b2all[tid] = b2;
  }
}

// ===========================================================================
// K1: implicit-GEMM 3x3 conv (K=576) + GELU + 1x1 expand to 128 (xa | k).
// One WG = 32 pixels of one image row. 8 waves: (mtile 0..3) x (pixgrp 0..1).
// ===========================================================================
__global__ void __launch_bounds__(256)
k1_conv(const float* __restrict__ x, const float* __restrict__ c1b,
        const float* __restrict__ c2b, const unsigned* __restrict__ afr,
        unsigned short* __restrict__ xaw, unsigned short* __restrict__ kw) {
  __shared__ __align__(32) unsigned xt[3 * 34 * 32];  // [row][col 0..33][chpair]
  __shared__ __align__(32) unsigned hst[32 * 32];     // [pix 0..31][chpair]
  int tid = threadIdx.x;
  int blk = blockIdx.x;
  int xseg = blk & 7, y = (blk >> 3) & 255, b = blk >> 11;
  int x0 = xseg * 32;

  // stage bf16 x-tile with 1-pixel halo, [row][col][ch] so B-frags are linear
  for (int i = tid; i < 3 * 32 * 34; i += 256) {
    int col = i % 34;
    int rem = i / 34;
    int cp = rem & 31, row = rem >> 5;
    int gx = x0 + col - 1, gy = y + row - 1;
    float lo = 0.f, hi = 0.f;
    if ((unsigned)gx < 256u && (unsigned)gy < 256u) {
      const float* p = x + (((size_t)(b * 64 + cp * 2)) << 16) + gy * 256 + gx;
      lo = p[0];
      hi = p[HWPIX];
    }
    xt[(row * 34 + col) * 32 + cp] = packbf(lo, hi);
  }
  __syncthreads();

  int lane = tid & 31, wave = tid >> 5;
  int lh = lane >> 4, ln = lane & 15;
  int m = wave & 3, pg = wave >> 2;
  int pixloc = pg * 16 + ln;

  v8f acc = {};
  for (int kc = 0; kc < 18; ++kc) {
    int tap = kc >> 1, dy = tap / 3, dx = tap - dy * 3;
    int cpb = ((kc & 1) << 4) + (lh << 3);  // chpair base for this lane half
    v16bf a = *(const v16bf*)(afr + (m * 18 + kc) * 256 + lane * 8);
    v16bf bb = *(const v16bf*)&xt[(dy * 34 + pixloc + dx) * 32 + cpb];
    acc = wmma_bf16(a, bb, acc);
  }
  // bias + exact GELU, stage h (bf16) to LDS [pix][chpair]
  int chb = m * 16 + lh * 8;
#pragma unroll
  for (int j = 0; j < 4; ++j) {
    float v0 = gelu_exact(acc[2 * j] + c1b[chb + 2 * j]);
    float v1 = gelu_exact(acc[2 * j + 1] + c1b[chb + 2 * j + 1]);
    hst[pixloc * 32 + (chb >> 1) + j] = packbf(v0, v1);
  }
  __syncthreads();

  // 64 -> 128 expand; each wave does 2 of the 16 (m2, pg2) tiles
  for (int t = 0; t < 2; ++t) {
    int m2 = wave, pg2 = t;
    int pl2 = pg2 * 16 + ln;
    v8f a2 = {};
    for (int kc = 0; kc < 2; ++kc) {
      v16bf af = *(const v16bf*)(afr + (72 + m2 * 2 + kc) * 256 + lane * 8);
      v16bf bf = *(const v16bf*)&hst[pl2 * 32 + kc * 16 + lh * 8];
      a2 = wmma_bf16(af, bf, a2);
    }
    int ch2 = m2 * 16 + lh * 8;
    uint4 pk;
    pk.x = packbf(a2[0] + c2b[ch2 + 0], a2[1] + c2b[ch2 + 1]);
    pk.y = packbf(a2[2] + c2b[ch2 + 2], a2[3] + c2b[ch2 + 3]);
    pk.z = packbf(a2[4] + c2b[ch2 + 4], a2[5] + c2b[ch2 + 5]);
    pk.w = packbf(a2[6] + c2b[ch2 + 6], a2[7] + c2b[ch2 + 7]);
    size_t pix = (((size_t)b) << 16) + (size_t)y * 256 + x0 + pl2;
    unsigned short* dstb = (m2 < 4) ? xaw : kw;
    int cho = (m2 < 4) ? ch2 : (ch2 - 64);
    *(uint4*)(dstb + pix * 64 + cho) = pk;   // 8 bf16 = 16B, coalesced
  }
}

// ===========================================================================
// K2: striped depthwise (1,3)+(3,1) + GELU, pixel-major bf16; block partial
// sums for the router pool (no atomics -> deterministic).
// ===========================================================================
__global__ void __launch_bounds__(256)
k2_dw(const unsigned short* __restrict__ xaw, const float* __restrict__ shw,
      const float* __restrict__ shb, const float* __restrict__ svw,
      const float* __restrict__ svb, unsigned short* __restrict__ xsw,
      float* __restrict__ part) {
  __shared__ float sd[256];
  int tid = threadIdx.x;
  int c = tid & 63, slot = tid >> 6;
  int blk = blockIdx.x;
  int b = blk >> 10, bi = blk & 1023;
  int px0 = bi * 64;
  float wh0 = shw[c * 3 + 0], wh1 = shw[c * 3 + 1], wh2 = shw[c * 3 + 2], bh = shb[c];
  float wv0 = svw[c * 3 + 0], wv1 = svw[c * 3 + 1], wv2 = svw[c * 3 + 2], bv = svb[c];
  const unsigned short* xab = xaw + (((size_t)b) << 16) * 64;
  float accsum = 0.f;
  for (int it = 0; it < 16; ++it) {
    int p = px0 + it * 4 + slot;
    int yy = p >> 8, xx = p & 255;
    float r = bv;
#pragma unroll
    for (int dy = 0; dy < 3; ++dy) {
      int y2 = yy + dy - 1;
      if ((unsigned)y2 < 256u) {
        float s = bh;
#pragma unroll
        for (int dx = 0; dx < 3; ++dx) {
          int x2 = xx + dx - 1;
          float xv = 0.f;
          if ((unsigned)x2 < 256u)
            xv = bf2f(xab[((size_t)(y2 * 256 + x2)) * 64 + c]);
          s += (dx == 0 ? wh0 : dx == 1 ? wh1 : wh2) * xv;
        }
        r += (dy == 0 ? wv0 : dy == 1 ? wv1 : wv2) * s;
      }
    }
    float xs = gelu_exact(r);
    xsw[((((size_t)b) << 16) + p) * 64 + c] = f2bf(xs);
    accsum += xs;
  }
  sd[tid] = accsum;
  __syncthreads();
  if (slot == 0) {
    float t = sd[c] + sd[64 + c] + sd[128 + c] + sd[192 + c];
    part[((size_t)(b * 1024 + bi)) * 64 + c] = t;
  }
}

// ===========================================================================
// K3: pooled -> logits -> fp32 softmax -> top-2 mask -> row-mask[16] + b3mix.
// ===========================================================================
__global__ void __launch_bounds__(256)
k3_route(const float* __restrict__ part, const float* __restrict__ gate,
         const float* __restrict__ e0b3, const float* __restrict__ e1b3,
         const float* __restrict__ e2b3, float* __restrict__ maskrow,
         float* __restrict__ b3mask) {
  __shared__ float sd[256];
  __shared__ float pooled[64];
  __shared__ float msk[3];
  int b = blockIdx.x, tid = threadIdx.x;
  int c = tid & 63, i0 = tid >> 6;
  float a = 0.f;
  for (int i = i0; i < 1024; i += 4) a += part[((size_t)(b * 1024 + i)) * 64 + c];
  sd[tid] = a;
  __syncthreads();
  if (tid < 64)
    pooled[c] = (sd[c] + sd[64 + c] + sd[128 + c] + sd[192 + c]) * (1.0f / 65536.0f);
  __syncthreads();
  if (tid == 0) {
    float lg[3];
    for (int e = 0; e < 3; ++e) {
      float s = 0.f;
      for (int cc = 0; cc < 64; ++cc) s += pooled[cc] * gate[e * 64 + cc];
      lg[e] = s;
    }
    float mx = fmaxf(lg[0], fmaxf(lg[1], lg[2]));
    float ex0 = expf(lg[0] - mx), ex1 = expf(lg[1] - mx), ex2 = expf(lg[2] - mx);
    float den = ex0 + ex1 + ex2;
    float w[3] = {ex0 / den, ex1 / den, ex2 / den};
    int i1 = 0;
    for (int e = 1; e < 3; ++e) if (w[e] > w[i1]) i1 = e;       // ties -> lower idx
    int i2 = -1;
    for (int e = 0; e < 3; ++e) {
      if (e == i1) continue;
      if (i2 < 0 || w[e] > w[i2]) i2 = e;
    }
    msk[0] = (i1 == 0 || i2 == 0) ? w[0] : 0.f;
    msk[1] = (i1 == 1 || i2 == 1) ? w[1] : 0.f;
    msk[2] = (i1 == 2 || i2 == 2) ? w[2] : 0.f;
  }
  __syncthreads();
  if (tid < 16)
    maskrow[b * 16 + tid] =
        tid < 2 ? msk[0] : tid < 6 ? msk[1] : tid < 14 ? msk[2] : 0.f;
  if (tid < 64)
    b3mask[b * 64 + c] = msk[0] * e0b3[c] + msk[1] * e1b3[c] + msk[2] * e2b3[c];
}

// ===========================================================================
// K4: MoE + proj.  Per wave: 16 pixels.
//   t1=W1all*xs, t2=W2all*k, u=(t1+b1)(t2+b2)*maskrow  (D-layout)
//   LDS round-trip u -> B-layout, e=W3all*u (4 mtiles)
//   out = xs + e + b3mix -> LDS -> proj B-frags -> proj WMMAs -> fp32 NCHW
// ===========================================================================
__global__ void __launch_bounds__(256)
k4_moe(const unsigned short* __restrict__ xsw, const unsigned short* __restrict__ kw,
       const unsigned* __restrict__ afr, const float* __restrict__ b1all,
       const float* __restrict__ b2all, const float* __restrict__ maskrow,
       const float* __restrict__ b3mask, const float* __restrict__ projb,
       float* __restrict__ out) {
  __shared__ __align__(32) unsigned ust[8][16 * 8];    // per wave: [px][rowpair]
  __shared__ __align__(32) unsigned ost[8][16 * 32];   // per wave: [px][chpair]
  int tid = threadIdx.x, lane = tid & 31, wave = tid >> 5;
  int lh = lane >> 4, ln = lane & 15;
  size_t pixg = (size_t)blockIdx.x * 128 + (size_t)wave * 16 + ln;
  int b = (int)(pixg >> 16);
  int pin = (int)(pixg & 65535);
  const unsigned short* xsp = xsw + pixg * 64;
  const unsigned short* kp = kw + pixg * 64;

  v16bf xsf[2], kf[2];
#pragma unroll
  for (int kc = 0; kc < 2; ++kc) {
    xsf[kc] = *(const v16bf*)(xsp + kc * 32 + lh * 16);
    kf[kc]  = *(const v16bf*)(kp  + kc * 32 + lh * 16);
  }

  v8f t1 = {}, t2 = {};
#pragma unroll
  for (int kc = 0; kc < 2; ++kc) {
    v16bf a1 = *(const v16bf*)(afr + (96 + kc) * 256 + lane * 8);
    t1 = wmma_bf16(a1, xsf[kc], t1);
    v16bf a2 = *(const v16bf*)(afr + (98 + kc) * 256 + lane * 8);
    t2 = wmma_bf16(a2, kf[kc], t2);
  }

  // u rows (D-layout: row = r + 8*lh), scaled by the router mask
#pragma unroll
  for (int j = 0; j < 4; ++j) {
    int r0 = lh * 8 + 2 * j, r1 = r0 + 1;
    float u0 = (t1[2 * j] + b1all[r0]) * (t2[2 * j] + b2all[r0]) * maskrow[b * 16 + r0];
    float u1 = (t1[2 * j + 1] + b1all[r1]) * (t2[2 * j + 1] + b2all[r1]) * maskrow[b * 16 + r1];
    ust[wave][ln * 8 + lh * 4 + j] = packbf(u0, u1);
  }
  __syncthreads();

  // u as B-matrix (K rows 0..15; lanes 16..31 hold the zero K=16..31 pad)
  v16bf uf = {};
  if (lh == 0) uf = *(const v16bf*)&ust[wave][ln * 8];

#pragma unroll
  for (int m = 0; m < 4; ++m) {
    v16bf a3 = *(const v16bf*)(afr + (100 + m) * 256 + lane * 8);
    v8f e = {};
    e = wmma_bf16(a3, uf, e);
    int chb = m * 16 + lh * 8;
    uint4 xr = *(const uint4*)(xsp + chb);        // xs in D-layout, 8 bf16
    const unsigned short* xh = (const unsigned short*)&xr;
#pragma unroll
    for (int j = 0; j < 4; ++j) {
      float o0 = bf2f(xh[2 * j]) + e[2 * j] + b3mask[b * 64 + chb + 2 * j];
      float o1 = bf2f(xh[2 * j + 1]) + e[2 * j + 1] + b3mask[b * 64 + chb + 2 * j + 1];
      ost[wave][ln * 32 + (chb >> 1) + j] = packbf(o0, o1);
    }
  }
  __syncthreads();

  v16bf obf[2];
#pragma unroll
  for (int kc = 0; kc < 2; ++kc)
    obf[kc] = *(const v16bf*)&ost[wave][ln * 32 + kc * 16 + lh * 8];

#pragma unroll
  for (int m = 0; m < 4; ++m) {
    v8f acc = {};
#pragma unroll
    for (int kc = 0; kc < 2; ++kc) {
      v16bf a = *(const v16bf*)(afr + (88 + m * 2 + kc) * 256 + lane * 8);
      acc = wmma_bf16(a, obf[kc], acc);
    }
    int chb = m * 16 + lh * 8;
#pragma unroll
    for (int r = 0; r < 8; ++r)
      out[(((size_t)(b * 64 + chb + r)) << 16) + pin] = acc[r] + projb[chb + r];
  }
}

// ===========================================================================
extern "C" void kernel_launch(void* const* d_in, const int* in_sizes, int n_in,
                              void* d_out, int out_size, void* d_ws, size_t ws_size,
                              hipStream_t stream) {
  (void)in_sizes; (void)n_in; (void)out_size; (void)ws_size;
  const float* x     = (const float*)d_in[0];
  const float* c1w   = (const float*)d_in[1];
  const float* c1b   = (const float*)d_in[2];
  const float* c2w   = (const float*)d_in[3];
  const float* c2b   = (const float*)d_in[4];
  const float* shw   = (const float*)d_in[5];
  const float* shb   = (const float*)d_in[6];
  const float* svw   = (const float*)d_in[7];
  const float* svb   = (const float*)d_in[8];
  const float* gate  = (const float*)d_in[9];
  const float* e0w1 = (const float*)d_in[10], *e0b1 = (const float*)d_in[11];
  const float* e0w2 = (const float*)d_in[12], *e0b2 = (const float*)d_in[13];
  const float* e0w3 = (const float*)d_in[14], *e0b3 = (const float*)d_in[15];
  const float* e1w1 = (const float*)d_in[16], *e1b1 = (const float*)d_in[17];
  const float* e1w2 = (const float*)d_in[18], *e1b2 = (const float*)d_in[19];
  const float* e1w3 = (const float*)d_in[20], *e1b3 = (const float*)d_in[21];
  const float* e2w1 = (const float*)d_in[22], *e2b1 = (const float*)d_in[23];
  const float* e2w2 = (const float*)d_in[24], *e2b2 = (const float*)d_in[25];
  const float* e2w3 = (const float*)d_in[26], *e2b3 = (const float*)d_in[27];
  const float* projw = (const float*)d_in[28];
  const float* projb = (const float*)d_in[29];

  char* ws = (char*)d_ws;
  unsigned* afr        = (unsigned*)(ws + AFRAG_OFF);
  float* b1all         = (float*)(ws + B1_OFF);
  float* b2all         = (float*)(ws + B2_OFF);
  float* maskrowp      = (float*)(ws + MASK_OFF);
  float* b3maskp       = (float*)(ws + B3M_OFF);
  unsigned short* xaw  = (unsigned short*)(ws + XA_OFF);
  unsigned short* kwp  = (unsigned short*)(ws + KW_OFF);
  unsigned short* xsw  = (unsigned short*)(ws + XS_OFF);
  float* part          = (float*)(ws + PART_OFF);

  k0_prep<<<1, 256, 0, stream>>>(c1w, c2w, projw,
                                 e0w1, e0b1, e0w2, e0b2, e0w3,
                                 e1w1, e1b1, e1w2, e1b2, e1w3,
                                 e2w1, e2b1, e2w2, e2b2, e2w3,
                                 afr, b1all, b2all);
  k1_conv<<<8192, 256, 0, stream>>>(x, c1b, c2b, afr, xaw, kwp);
  k2_dw<<<4096, 256, 0, stream>>>(xaw, shw, shb, svw, svb, xsw, part);
  k3_route<<<4, 256, 0, stream>>>(part, gate, e0b3, e1b3, e2b3, maskrowp, b3maskp);
  k4_moe<<<2048, 256, 0, stream>>>(xsw, kwp, afr, b1all, b2all, maskrowp, b3maskp,
                                   projb, (float*)d_out);
}
